// NNaNUnit_57921928954094
// MI455X (gfx1250) — matrix operand order repaired and checked by reading, not compiled, not run on hardware
//
#include <hip/hip_runtime.h>

// Per-element MLP (1->8->8->1, relu, +residual) over 33.5M fp32 elements.
// Memory-bound: 268 MB @ 23.3 TB/s ~= 11.5 us floor; ~165 FLOP/elem would be
// VALU-compute-bound, so layer GEMMs go through WMMA:
//   layer0: v_wmma_f32_16x16x4_f32   (fp32 outer product, 32 elems/wmma)
//   layer1: v_wmma_f32_16x16x32_f16  (block-diagonal W1^T, 32 elems/wmma)
//   layer2: v_wmma_f32_16x16x32_f16  (W2 rows 0/8, result in D-VGPR0 all lanes)
// Elements ride the N axis (lane = element), features the M axis, with two
// 16-element groups on disjoint K ranges matching the wave32 lane-half layout.
// Relu is applied in packed f16 (v_cvt_pk_f16_f32 + v_pk_max_num_f16).
// The four float4 sub-pipelines are written STAGE-MAJOR so the scheduler can
// fill the WMMA->VALU RAW hazard slots (4 co-exec NOPs for *F16 WMMA) with
// the independent sibling pipelines instead of v_nop.

typedef __attribute__((ext_vector_type(16))) _Float16 v16h;
typedef __attribute__((ext_vector_type(8)))  float    v8f;
typedef __attribute__((ext_vector_type(4)))  float    v4f;
typedef __attribute__((ext_vector_type(2)))  float    v2f;

__device__ __forceinline__ v16h relu_pack_f16(v8f d) {
  v16h b;
#pragma unroll
  for (int i = 0; i < 8; ++i) b[i] = (_Float16)d[i];   // fuses to v_cvt_pk_f16_f32
#pragma unroll
  for (int i = 8; i < 16; ++i) b[i] = (_Float16)0.0f;
  v16h z;
#pragma unroll
  for (int i = 0; i < 16; ++i) z[i] = (_Float16)0.0f;
  // relu in f16: monotone conversion commutes with max(.,0) -> v_pk_max_num_f16
  return __builtin_elementwise_max(b, z);
}

__global__ __launch_bounds__(256) void mlp_wmma_main(
    const float* __restrict__ x,
    const float* __restrict__ W0, const float* __restrict__ Bi0,
    const float* __restrict__ W1, const float* __restrict__ Bi1,
    const float* __restrict__ W2, const float* __restrict__ Bi2,
    float* __restrict__ out, long long n_full) {
  const int lane = (int)(threadIdx.x & 31u);

  // ---- wave-constant WMMA operands (built once; weights are tiny & L2-hot) ----
  // Layer0 A (f32 16x4): row m<8 -> W0[m] at K=0 (lanes 0-7, VGPR0);
  //                      row m>=8 -> W0[m-8] at K=2 (lanes 24-31, VGPR0).
  v2f a0; a0[0] = 0.0f; a0[1] = 0.0f;
  if (lane < 8)        a0[0] = W0[lane];
  else if (lane >= 24) a0[0] = W0[lane - 24];

  // Layer1 A (f16 16x32): block-diagonal W1^T.
  //   rows 0-7  : A[m][k]    = W1[k][m]   for k=0..7   -> lane m,   halves 0-7
  //   rows 8-15 : W1[k][m-8] for k=0..7   -> lane m+16, halves 0-7
  //   lanes 8-23 stay zero.
  v16h a1;
#pragma unroll
  for (int i = 0; i < 16; ++i) a1[i] = (_Float16)0.0f;
  if (lane < 8) {
#pragma unroll
    for (int k = 0; k < 8; ++k) a1[k] = (_Float16)W1[k * 8 + lane];
  } else if (lane >= 24) {
#pragma unroll
    for (int k = 0; k < 8; ++k) a1[k] = (_Float16)W1[k * 8 + (lane - 24)];
  }

  // Layer2 A (f16 16x32): row 0 -> W2 in lane 0 elems 0..7 ; row 8 -> W2 in lane 24 elems 0..7.
  v16h a2;
#pragma unroll
  for (int i = 0; i < 16; ++i) a2[i] = (_Float16)0.0f;
  if (lane == 0) {
#pragma unroll
    for (int k = 0; k < 8; ++k) a2[k] = (_Float16)W2[k];
  } else if (lane == 24) {
#pragma unroll
    for (int k = 0; k < 8; ++k) a2[k] = (_Float16)W2[k];
  }

  // Bias accumulators: C[m][n] = bias[m%8] -> c[v] = bias[v] for every lane.
  v8f c0, c1, c2;
#pragma unroll
  for (int v = 0; v < 8; ++v) { c0[v] = Bi0[v]; c1[v] = Bi1[v]; c2[v] = Bi2[0]; }

  const long long wave = (((long long)blockIdx.x * (long long)blockDim.x) +
                          (long long)threadIdx.x) >> 5;
  const long long base = wave * 128;
  if (base + 128 > n_full) return;  // wave-uniform: EXEC stays all-ones

  // Streaming float4 per lane (nontemporal: read-once / write-once data).
  v4f xv = __builtin_nontemporal_load((const v4f*)(x + base) + lane);

  // ---- stage-major pipeline over the 4 independent element sets ----
  v8f d0[4];
#pragma unroll
  for (int j = 0; j < 4; ++j) {
    // Layer0 B (f32 4x16): VGPR0 = x[lane] (K=0 for lanes 0-15, K=2 for 16-31).
    v2f bb; bb[0] = xv[j]; bb[1] = 0.0f;
    d0[j] = __builtin_amdgcn_wmma_f32_16x16x4_f32(
        false, a0, false, bb, (short)0, c0, false, false);
  }

  v16h h1[4];
#pragma unroll
  for (int j = 0; j < 4; ++j) h1[j] = relu_pack_f16(d0[j]);

  v8f d1[4];
#pragma unroll
  for (int j = 0; j < 4; ++j)
    d1[j] = __builtin_amdgcn_wmma_f32_16x16x32_f16(
        false, a1, false, h1[j], (short)0, c1, false, false);

  v16h h2[4];
#pragma unroll
  for (int j = 0; j < 4; ++j) h2[j] = relu_pack_f16(d1[j]);

  v8f d2[4];
#pragma unroll
  for (int j = 0; j < 4; ++j)
    d2[j] = __builtin_amdgcn_wmma_f32_16x16x32_f16(
        false, a2, false, h2[j], (short)0, c2, false, false);

  v4f ov;
#pragma unroll
  for (int j = 0; j < 4; ++j) ov[j] = d2[j][0] + xv[j];  // residual; y is in D-VGPR0

  __builtin_nontemporal_store(ov, (v4f*)(out + base) + lane);
}

// Scalar tail for the (rare) remainder; pure fp32 VALU with bounds guard.
__global__ __launch_bounds__(256) void mlp_tail(
    const float* __restrict__ x,
    const float* __restrict__ W0, const float* __restrict__ Bi0,
    const float* __restrict__ W1, const float* __restrict__ Bi1,
    const float* __restrict__ W2, const float* __restrict__ Bi2,
    float* __restrict__ out, long long start, long long n) {
  long long i = start + (long long)blockIdx.x * blockDim.x + threadIdx.x;
  if (i >= n) return;
  float xi = x[i];
  float h[8];
#pragma unroll
  for (int m = 0; m < 8; ++m) {
    float t = xi * W0[m] + Bi0[m];
    h[m] = t > 0.0f ? t : 0.0f;
  }
  float g[8];
#pragma unroll
  for (int m = 0; m < 8; ++m) {
    float s = Bi1[m];
#pragma unroll
    for (int k = 0; k < 8; ++k) s = fmaf(h[k], W1[k * 8 + m], s);
    g[m] = s > 0.0f ? s : 0.0f;
  }
  float y = Bi2[0];
#pragma unroll
  for (int k = 0; k < 8; ++k) y = fmaf(g[k], W2[k], y);
  out[i] = y + xi;
}

extern "C" void kernel_launch(void* const* d_in, const int* in_sizes, int n_in,
                              void* d_out, int out_size, void* d_ws, size_t ws_size,
                              hipStream_t stream) {
  const float* x  = (const float*)d_in[0];
  const float* W0 = (const float*)d_in[1];
  const float* b0 = (const float*)d_in[2];
  const float* W1 = (const float*)d_in[3];
  const float* b1 = (const float*)d_in[4];
  const float* W2 = (const float*)d_in[5];
  const float* b2 = (const float*)d_in[6];
  float* out = (float*)d_out;

  const long long N = (long long)in_sizes[0];
  const long long n_full = (N / 128) * 128;        // 128 elements per wave tile
  const long long waves  = n_full / 128;

  if (waves > 0) {
    const long long threads = waves * 32;
    const int block = 256;
    const long long blocks = (threads + block - 1) / block;
    mlp_wmma_main<<<(unsigned)blocks, block, 0, stream>>>(
        x, W0, b0, W1, b1, W2, b2, out, n_full);
  }
  const long long rem = N - n_full;
  if (rem > 0) {
    const int block = 256;
    const long long blocks = (rem + block - 1) / block;
    mlp_tail<<<(unsigned)blocks, block, 0, stream>>>(
        x, W0, b0, W1, b1, W2, b2, out, n_full, N);
  }
}
